// SpiralDeblock_9526237463032
// MI455X (gfx1250) — matrix operand order repaired and checked by reading, not compiled, not run on hardware
//
#include <hip/hip_runtime.h>

typedef _Float16 v16h __attribute__((ext_vector_type(16)));
typedef _Float16 v8h  __attribute__((ext_vector_type(8)));
typedef _Float16 v4h  __attribute__((ext_vector_type(4)));
typedef float    v8f  __attribute__((ext_vector_type(8)));

#define B_      64
#define N_IN    195
#define N_OUT   778
#define S_      9
#define D_IN    256
#define K_      2304                 // S_ * D_IN
#define D_OUT   128
#define M_TOT   (B_ * N_OUT)         // 49792 (divisible by 16)
#define KPAD    2312                 // K_ + 8 halfs pad: row stride 1156 dw, 1156%64=4 -> conflict-free
#define LDS_A_BYTES (16 * KPAD * 2)  // 73984
#define LDS_COL_OFF LDS_A_BYTES
#define LDS_VAL_OFF (LDS_A_BYTES + 144 * 3 * 4)
#define LDS_TOTAL   (LDS_A_BYTES + 2 * 144 * 3 * 4)  // 77440 bytes

// --- Prep: Wt[n][k] = (f16) W[k][n]  (column-major over K so B-fragments are contiguous) ---
__global__ __launch_bounds__(256)
void wt_transpose_kernel(const float* __restrict__ W, _Float16* __restrict__ Wt) {
    int i = blockIdx.x * 256 + threadIdx.x;          // over 128*2304
    if (i < D_OUT * K_) {
        int n = i / K_;
        int k = i - n * K_;
        Wt[i] = (_Float16)W[(size_t)k * D_OUT + n];
    }
}

// --- Fused pool + spiral-gather + WMMA GEMM + bias + ReLU ---
__global__ __launch_bounds__(256)
void spiral_gemm_kernel(const float* __restrict__ x,
                        const int* __restrict__ col,
                        const float* __restrict__ value,
                        const int* __restrict__ indices,
                        const _Float16* __restrict__ Wt,
                        const float* __restrict__ bias,
                        float* __restrict__ out) {
    extern __shared__ unsigned char smem[];
    _Float16* sA  = (_Float16*)smem;
    int*      sCol = (int*)(smem + LDS_COL_OFF);
    float*    sVal = (float*)(smem + LDS_VAL_OFF);

    const int t  = threadIdx.x;
    const int m0 = blockIdx.x * 16;                  // first output row of this tile

    // Phase 0: resolve (row r, spiral s) -> 3 weighted x-row sources (144 tasks)
    if (t < 144) {
        int r  = t / S_;
        int s  = t - r * S_;
        int m  = m0 + r;
        int bb = m / N_OUT;
        int nn = m - bb * N_OUT;
        int idx = indices[nn * S_ + s];
        int j   = idx * 3;                           // row = repeat(arange(N_out), 3)
        #pragma unroll
        for (int k = 0; k < 3; ++k) {
            int c = col[j + k];
            sCol[t * 3 + k] = (bb * N_IN + c) * (D_IN / 4);  // in float4 units
            sVal[t * 3 + k] = value[j + k];
        }
    }
    __syncthreads();

    // Phase 1: pool 3 weighted rows, convert to f16, stage 16 x 2304 A-tile in LDS
    const float4* x4 = (const float4*)x;
    for (int i = t; i < 144 * 64; i += 256) {        // 144 tasks * 64 float4 each
        int task = i >> 6;
        int q    = i & 63;
        int o0 = sCol[task * 3 + 0], o1 = sCol[task * 3 + 1], o2 = sCol[task * 3 + 2];
        float v0 = sVal[task * 3 + 0], v1 = sVal[task * 3 + 1], v2 = sVal[task * 3 + 2];
        float4 a0 = x4[o0 + q];
        float4 a1 = x4[o1 + q];
        float4 a2 = x4[o2 + q];
        float px = fmaf(v2, a2.x, fmaf(v1, a1.x, v0 * a0.x));
        float py = fmaf(v2, a2.y, fmaf(v1, a1.y, v0 * a0.y));
        float pz = fmaf(v2, a2.z, fmaf(v1, a1.z, v0 * a0.z));
        float pw = fmaf(v2, a2.w, fmaf(v1, a1.w, v0 * a0.w));
        v4h hv = { (_Float16)px, (_Float16)py, (_Float16)pz, (_Float16)pw };
        int r = task / S_;
        int s = task - r * S_;
        *(v4h*)&sA[r * KPAD + s * D_IN + (q << 2)] = hv;
    }
    __syncthreads();

    // Phase 2: each wave computes one 16x16 output tile; K loop of v_wmma_f32_16x16x32_f16
    const int lane = t & 31;
    const int wv   = t >> 5;                          // wave 0..7 -> N columns [16w, 16w+16)
    const int colN = (wv << 4) + (lane & 15);
    const int hiH  = lane >> 4;                       // 0: lanes 0-15, 1: lanes 16-31
    // A-frag (16-bit 16x32): lane<16 row=lane, K {0..7,16..23}; lane>=16 row=lane-16, K {8..15,24..31}
    const _Float16* arow = sA + (lane & 15) * KPAD + (hiH << 3);
    // B-frag (32x16): col = lane%16, K {0..7,16..23} for lanes 0-15, {8..15,24..31} for lanes 16-31 (matches A)
    const _Float16* wcol = Wt + (size_t)colN * K_ + (hiH << 3);

    v8f acc = {};
    #pragma unroll 4
    for (int k0 = 0; k0 < K_; k0 += 32) {
        v8h lo = *(const v8h*)(arow + k0);            // K chunk +0..7   (16B ds_load)
        v8h hi = *(const v8h*)(arow + k0 + 16);       // K chunk +16..23 (16B ds_load)
        v16h a;
        #pragma unroll
        for (int e = 0; e < 8; ++e) { a[e] = lo[e]; a[e + 8] = hi[e]; }
        v8h blo = *(const v8h*)(wcol + k0);
        v8h bhi = *(const v8h*)(wcol + k0 + 16);
        v16h bm;
        #pragma unroll
        for (int e = 0; e < 8; ++e) { bm[e] = blo[e]; bm[e + 8] = bhi[e]; }
        acc = __builtin_amdgcn_wmma_f32_16x16x32_f16(
            /*neg_a=*/false, a, /*neg_b=*/false, bm,
            /*c_mod=*/(short)0, acc, /*reuse_a=*/false, /*reuse_b=*/false);
    }

    // Epilogue: C/D layout — VGPR j: lanes 0-15 -> M=j, lanes 16-31 -> M=8+j; N = lane%16
    const float bb = bias[colN];
    const int baseRow = m0 + (hiH << 3);
    #pragma unroll
    for (int j = 0; j < 8; ++j) {
        float v = acc[j] + bb;
        out[(size_t)(baseRow + j) * D_OUT + colN] = v > 0.f ? v : 0.f;
    }
}

extern "C" void kernel_launch(void* const* d_in, const int* in_sizes, int n_in,
                              void* d_out, int out_size, void* d_ws, size_t ws_size,
                              hipStream_t stream) {
    // setup_inputs order: x, row, col, value, indices, W, b
    const float* x       = (const float*)d_in[0];
    // d_in[1] (row) is structurally repeat(arange(N_out), 3) — consumed implicitly
    const int*   col     = (const int*)d_in[2];
    const float* value   = (const float*)d_in[3];
    const int*   indices = (const int*)d_in[4];
    const float* W       = (const float*)d_in[5];
    const float* bias    = (const float*)d_in[6];
    float*       out     = (float*)d_out;

    _Float16* Wt = (_Float16*)d_ws;                   // 2304*128*2 = 589,824 bytes

    (void)in_sizes; (void)n_in; (void)out_size; (void)ws_size;

    wt_transpose_kernel<<<(D_OUT * K_ + 255) / 256, 256, 0, stream>>>(W, Wt);
    spiral_gemm_kernel<<<M_TOT / 16, 256, LDS_TOTAL, stream>>>(
        x, col, value, indices, Wt, bias, out);
}
